// RuleFocusedAttention_1597727834391
// MI455X (gfx1250) — matrix operand order repaired and checked by reading, not compiled, not run on hardware
//
#include <hip/hip_runtime.h>
#include <math.h>

// Problem constants (match reference)
#define N_TOT 262144
#define DD 128

typedef __attribute__((ext_vector_type(16))) __bf16 v16bf;
typedef __attribute__((ext_vector_type(8)))  float  v8f;

// ---- bf16 helpers (round-to-nearest-even) ----
__device__ __forceinline__ unsigned short f2bf(float f) {
  unsigned int u = __float_as_uint(f);
  unsigned int r = (u + 0x7FFFu + ((u >> 16) & 1u)) >> 16;
  return (unsigned short)r;
}
__device__ __forceinline__ float bf2f(unsigned short h) {
  return __uint_as_float(((unsigned int)h) << 16);
}
__device__ __forceinline__ unsigned int packbf(float a, float b) {
  return (unsigned int)f2bf(a) | ((unsigned int)f2bf(b) << 16);
}

// Workspace layout (uint32/float units):
//   [0..127]    sum_light[c]
//   [128..255]  sum_stop[c]
//   [256]       cnt_light   [257] cnt_stop
//   [258..385]  cvec[j] = b1[j] + W1b.h_light + W1c.h_stop
//   [512..8703]    W1a bf16 HI, packed in WMMA-B lane layout (8192 words)
//   [8704..16895]  W1a bf16 LO (8192 words)
#define WS_CNT   256
#define WS_CVEC  258
#define WS_WHI   512
#define WS_WLO   8704

// ------------------- Kernel 1: masked pooling sums -------------------
__global__ __launch_bounds__(256) void k_pool(const float* __restrict__ h,
                                              const int* __restrict__ types,
                                              const unsigned char* __restrict__ masks,
                                              float* __restrict__ ws) {
  __shared__ float red[256];
  __shared__ float redc[4];
  const int tid  = threadIdx.x;
  const int c    = tid & 127;
  const int half = tid >> 7;

  float aL = 0.f, aS = 0.f, cL = 0.f, cS = 0.f;
  for (int base = blockIdx.x * 64; base < N_TOT; base += gridDim.x * 64) {
    for (int rr = half; rr < 64; rr += 2) {
      const int i = base + rr;
      const int t = types[i];
      const bool m = masks[i] != 0;
      const float v = h[(size_t)i * DD + c];
      if (m && t == 1) { aL += v; cL += 1.f; }
      if (m && t == 2) { aS += v; cS += 1.f; }
    }
  }
  red[tid] = aL;
  if (c == 0) { redc[half] = cL; redc[2 + half] = cS; }
  __syncthreads();
  float tL = 0.f;
  if (half == 0) tL = aL + red[tid + 128];
  __syncthreads();
  red[tid] = aS;
  __syncthreads();
  if (half == 0) {
    const float tS = aS + red[tid + 128];
    atomicAdd(&ws[c], tL);
    atomicAdd(&ws[128 + c], tS);
  }
  if (tid == 0) {
    atomicAdd(&ws[WS_CNT + 0], redc[0] + redc[1]);
    atomicAdd(&ws[WS_CNT + 1], redc[2] + redc[3]);
  }
}

// ------------------- Kernel 2: means, fused bias c, W1a bf16 packing -------------------
__global__ __launch_bounds__(256) void k_prep(const float* __restrict__ W1,
                                              const float* __restrict__ b1,
                                              float* __restrict__ ws) {
  __shared__ float sL[DD], sS[DD];
  const int tid = threadIdx.x;
  if (tid < DD) {
    const float cntL = ws[WS_CNT + 0], cntS = ws[WS_CNT + 1];
    sL[tid] = (cntL > 0.f) ? ws[tid]        / fmaxf(cntL, 1.f) : 0.f;
    sS[tid] = (cntS > 0.f) ? ws[128 + tid]  / fmaxf(cntS, 1.f) : 0.f;
  }
  __syncthreads();
  if (tid < DD) {
    float acc = b1[tid];
    const float* row = W1 + (size_t)tid * (3 * DD);
    for (int k = 0; k < DD; ++k)
      acc += sL[k] * row[DD + k] + sS[k] * row[2 * DD + k];
    ws[WS_CVEC + tid] = acc;
  }
  // Pack W1a (= W1[:, 0:128]) into bf16 hi/lo for the WMMA B (32x16 K-major) operand in this code's contiguous k order (same as A):
  // item = ((nt*4 + kt)*32 + lane)*8 + w ;  lane<16 holds K=kt*32+[0..15], lane>=16 K=kt*32+16+[0..15]
  unsigned int* whi = (unsigned int*)ws + WS_WHI;
  unsigned int* wlo = (unsigned int*)ws + WS_WLO;
  for (int item = tid; item < 8192; item += 256) {
    const int w    = item & 7;
    const int lane = (item >> 3) & 31;
    const int kt   = (item >> 8) & 3;
    const int nt   = item >> 10;
    const int j = nt * 16 + (lane & 15);           // output-unit (N) index
    const int K = kt * 32 + (lane >> 4) * 16 + w * 2;
    const float x0 = W1[(size_t)j * (3 * DD) + K];
    const float x1 = W1[(size_t)j * (3 * DD) + K + 1];
    const unsigned short h0 = f2bf(x0), h1 = f2bf(x1);
    whi[item] = (unsigned int)h0 | ((unsigned int)h1 << 16);
    wlo[item] = packbf(x0 - bf2f(h0), x1 - bf2f(h1));
  }
}

// ------------------- Kernel 3: WMMA GEMM + fused epilogue -------------------
__global__ __launch_bounds__(256) void k_main(const float* __restrict__ h,
                                              const int* __restrict__ types,
                                              const unsigned char* __restrict__ masks,
                                              const int* __restrict__ rule_id,
                                              const float* __restrict__ W2,
                                              const float* __restrict__ b2,
                                              const float* __restrict__ rule_table,
                                              const float* __restrict__ ws,
                                              float* __restrict__ hout,
                                              float* __restrict__ bout) {
  __shared__ unsigned int sW[16384];   // 64KB: [0..8191]=hi, [8192..16383]=lo
  __shared__ float sC[DD];
  __shared__ float sW2[DD];

  const int tid = threadIdx.x;
  {  // stage packed W into LDS (b128 copies)
    const uint4* src = (const uint4*)((const unsigned int*)ws + WS_WHI);
    uint4* dst = (uint4*)sW;
    for (int idx = tid; idx < 4096; idx += 256) dst[idx] = src[idx];
  }
  if (tid < DD) { sC[tid] = ws[WS_CVEC + tid]; sW2[tid] = W2[tid]; }
  __syncthreads();

  const int lane  = tid & 31;
  const int wave  = tid >> 5;
  const int mtile = (blockIdx.x * 8 + wave) * 16;
  const int mrow  = lane & 15;   // A-matrix M = lane%16
  const int khalf = lane >> 4;   // A-matrix K-half selector
  const int colb  = (lane & 15) * 8;

  // per-lane rule embedding slice + scalar bias
  const int rid = rule_id[0];
  float remb[8];
#pragma unroll
  for (int e = 0; e < 8; ++e) remb[e] = rule_table[(size_t)rid * DD + colb + e];
  const float b2v = b2[0];

  union U16 { unsigned int u[8]; uint4 q[2]; v16bf v; };

  v8f acc[8];
  const v8f vzero = {0.f,0.f,0.f,0.f,0.f,0.f,0.f,0.f};
#pragma unroll
  for (int nt = 0; nt < 8; ++nt) acc[nt] = vzero;

  const float* arow = h + (size_t)(mtile + mrow) * DD;

#pragma unroll
  for (int kb = 0; kb < 4; ++kb) {
    // A 16x32 bf16 in this code's contiguous k order (matches B packing): lane<16 holds K = kb*32 + [0..15] (v0-7);
    // lane>=16 holds +16..31  -> two contiguous 8-float chunks per lane.
    const float4* p0 = (const float4*)(arow + kb * 32 + khalf * 16);
    const float4* p1 = (const float4*)(arow + kb * 32 + khalf * 16 + 8);
    const float4 a0 = p0[0], a1 = p0[1];
    const float4 b0 = p1[0], b1v = p1[1];
    const float c0[8] = {a0.x,a0.y,a0.z,a0.w,a1.x,a1.y,a1.z,a1.w};
    const float c1[8] = {b0.x,b0.y,b0.z,b0.w,b1v.x,b1v.y,b1v.z,b1v.w};
    U16 aHi, aLo;
#pragma unroll
    for (int w = 0; w < 4; ++w) {
      const float x0 = c0[2*w], x1 = c0[2*w+1];
      const unsigned short h0 = f2bf(x0), h1 = f2bf(x1);
      aHi.u[w]     = (unsigned int)h0 | ((unsigned int)h1 << 16);
      aLo.u[w]     = packbf(x0 - bf2f(h0), x1 - bf2f(h1));
      const float y0 = c1[2*w], y1 = c1[2*w+1];
      const unsigned short g0 = f2bf(y0), g1 = f2bf(y1);
      aHi.u[4 + w] = (unsigned int)g0 | ((unsigned int)g1 << 16);
      aLo.u[4 + w] = packbf(y0 - bf2f(g0), y1 - bf2f(g1));
    }
#pragma unroll
    for (int nt = 0; nt < 8; ++nt) {
      U16 bHi, bLo;
      const int base = ((nt * 4 + kb) * 32 + lane) * 8;
      const uint4* ph = (const uint4*)&sW[base];
      const uint4* pl = (const uint4*)&sW[8192 + base];
      bHi.q[0] = ph[0]; bHi.q[1] = ph[1];
      bLo.q[0] = pl[0]; bLo.q[1] = pl[1];
      // bf16x3 split-precision accumulation (hi*hi + lo*hi + hi*lo), f32 accum
      acc[nt] = __builtin_amdgcn_wmma_f32_16x16x32_bf16(false, aHi.v, false, bHi.v,
                                                        (short)0, acc[nt], false, false);
      acc[nt] = __builtin_amdgcn_wmma_f32_16x16x32_bf16(false, aLo.v, false, bHi.v,
                                                        (short)0, acc[nt], false, false);
      acc[nt] = __builtin_amdgcn_wmma_f32_16x16x32_bf16(false, aHi.v, false, bLo.v,
                                                        (short)0, acc[nt], false, false);
    }
  }

  // Epilogue: hid = relu(acc + c); partial dot with w2; reduce across the 16
  // lanes of each half (C/D layout: VGPR r -> M = r + (lane/16)*8, N = lane%16)
  const int col = lane & 15;
  float betas[8];
#pragma unroll
  for (int r = 0; r < 8; ++r) {
    float p = 0.f;
#pragma unroll
    for (int nt = 0; nt < 8; ++nt) {
      const int j = nt * 16 + col;
      const float hv = fmaxf(acc[nt][r] + sC[j], 0.f);
      p += hv * sW2[j];
    }
    p += __shfl_xor(p, 1);
    p += __shfl_xor(p, 2);
    p += __shfl_xor(p, 4);
    p += __shfl_xor(p, 8);
    betas[r] = 1.f / (1.f + expf(-(p + b2v)));
  }

  // Blend + write: lanes 0-15 own rows mtile+[0..7], lanes 16-31 rows mtile+8+[0..7];
  // each lane writes 8 contiguous cols (two b128 stores per row).
  const int ohalf = lane >> 4;
#pragma unroll
  for (int r = 0; r < 8; ++r) {
    const int row = mtile + ohalf * 8 + r;
    const float beta = betas[r];
    const int t = types[row];
    const bool car = (t == 0) && (masks[row] != 0);
    const float* hr = h + (size_t)row * DD + colb;
    const float4 x0 = ((const float4*)hr)[0];
    const float4 x1 = ((const float4*)hr)[1];
    const float bb  = car ? beta : 1.f;
    const float omb = car ? (1.f - beta) : 0.f;
    float4 o0, o1;
    o0.x = bb * x0.x + omb * remb[0];
    o0.y = bb * x0.y + omb * remb[1];
    o0.z = bb * x0.z + omb * remb[2];
    o0.w = bb * x0.w + omb * remb[3];
    o1.x = bb * x1.x + omb * remb[4];
    o1.y = bb * x1.y + omb * remb[5];
    o1.z = bb * x1.z + omb * remb[6];
    o1.w = bb * x1.w + omb * remb[7];
    float* orow = hout + (size_t)row * DD + colb;
    ((float4*)orow)[0] = o0;
    ((float4*)orow)[1] = o1;
    if (col == 0) bout[row] = car ? beta : 0.f;
  }
}

// ------------------- Launch -------------------
extern "C" void kernel_launch(void* const* d_in, const int* in_sizes, int n_in,
                              void* d_out, int out_size, void* d_ws, size_t ws_size,
                              hipStream_t stream) {
  const float*         h     = (const float*)d_in[0];
  const int*           types = (const int*)d_in[1];
  const unsigned char* masks = (const unsigned char*)d_in[2];
  const int*           rid   = (const int*)d_in[3];
  const float*         W1    = (const float*)d_in[4];
  const float*         b1    = (const float*)d_in[5];
  const float*         W2    = (const float*)d_in[6];
  const float*         b2    = (const float*)d_in[7];
  const float*         rtab  = (const float*)d_in[8];

  float* ws   = (float*)d_ws;
  float* hout = (float*)d_out;
  float* bout = hout + (size_t)N_TOT * DD;

  // zero the atomic accumulators (sums + counts) every call
  hipMemsetAsync(d_ws, 0, (size_t)(WS_CNT + 2) * sizeof(float), stream);

  k_pool<<<512, 256, 0, stream>>>(h, types, masks, ws);
  k_prep<<<1, 256, 0, stream>>>(W1, b1, ws);
  k_main<<<N_TOT / 128, 256, 0, stream>>>(h, types, masks, rid, W2, b2, rtab, ws,
                                          hout, bout);
}
